// DGCNN_58368605553186
// MI455X (gfx1250) — matrix operand order repaired and checked by reading, not compiled, not run on hardware
//
#include <hip/hip_runtime.h>
#include <hip/hip_bf16.h>
#include <math.h>

// ---------------------------------------------------------------------------
// DGCNN normal-estimation forward for MI455X (gfx1250, wave32, WMMA).
// All 1x1 convs run as f16 WMMA GEMMs with f32 accumulate. Edge features,
// BN+LeakyReLU epilogues and k/N max-pools are fused so no huge intermediate
// ([B,2C,N,k] or [B,1024,N]) ever touches HBM. Weights are converted once to
// f16 in WMMA-fragment order, so all LDS weight staging is linear 32B copies;
// operand loads are 2x ds_load_b128 per lane. Activation staging is
// register-batched with compile-time trip counts; GEMMs double-buffer LDS.
// ---------------------------------------------------------------------------

typedef _Float16 f16;
typedef __attribute__((ext_vector_type(16))) _Float16 v16h;
typedef __attribute__((ext_vector_type(8)))  float    v8f;

#define KNN  40
#define FRAG 512  // f16 elements per 16x32 / 32x16 fragment (1 KB)

static __device__ __forceinline__ float lrelu(float y) {
  return y >= 0.f ? y : 0.2f * y;
}
static __device__ __forceinline__ float bn_apply(float y, const float* g, const float* b,
                                                 const float* m, const float* v, int o) {
  float s = g[o] * rsqrtf(v[o] + 1e-5f);
  return s * (y - m[o]) + b[o];
}
static __device__ __forceinline__ v8f wmma_f16(v16h a, v16h b, v8f c) {
  return __builtin_amdgcn_wmma_f32_16x16x32_f16(false, a, false, b, (short)0, c, false, false);
}

// Fragment-order position of element (m,k) of a 16x32 A-tile.
// Forward layout: lane = m + 16*g, element j: k = 16*(j>=8) + 8*g + 2*((j>>1)&3) + (j&1)
static __device__ __forceinline__ int fragA_pos(int m, int k) {
  int lane = m + 16 * ((k >> 3) & 1);
  int i = 4 * ((k >> 4) & 1) + ((k >> 1) & 3);
  return lane * 16 + 2 * i + (k & 1);
}
// Fragment-order position of element (k,n) of a 32x16 B-tile.
// Forward layout: lane = n + 16*g, element j: k = 16*(j>=8) + 8*g + (j&7)
static __device__ __forceinline__ int fragB_pos(int k, int n) {
  int lane = n + 16 * ((k >> 3) & 1);
  return lane * 16 + 8 * ((k >> 4) & 1) + (k & 7);
}
// One aligned 32-byte LDS read per lane (lowered to 2x ds_load_b128).
static __device__ __forceinline__ v16h frag_load(const f16* s, int lane) {
  return *(const v16h*)(s + lane * 16);
}

// ---------------------------------------------------------------------------
// One-time weight prep: [O, Cin] f32 row-major -> f16 fragment-tile order with
// K padded to KP (zeros). Tile (o/16, c/32) stored at ((o/16)*(KP/32)+(c/32)).
// ---------------------------------------------------------------------------
__global__ void prep_weights_kernel(const float* __restrict__ W, int O, int Cin, int KP,
                                    f16* __restrict__ dst) {
  int t = blockIdx.x * blockDim.x + threadIdx.x;
  if (t >= O * KP) return;
  int o = t / KP, c = t % KP;
  float v = (c < Cin) ? W[(size_t)o * Cin + c] : 0.f;
  dst[((size_t)((o >> 4) * (KP >> 5) + (c >> 5))) * FRAG + fragA_pos(o & 15, c & 31)] =
      (f16)v;
}

// ---------------------------------------------------------------------------
// kNN: per point, top-40 by -squared-distance (insertion selection).
// x[b*xbs + c*ld + n]; C is compile-time so the distance loop fully unrolls.
// ---------------------------------------------------------------------------
template <int C>
__global__ void knn_kernel(const float* __restrict__ x, long xbs, int ld,
                           int* __restrict__ idx, int N) {
  int n = blockIdx.x * blockDim.x + threadIdx.x;
  int b = blockIdx.y;
  if (n >= N) return;
  const float* xb = x + (size_t)b * xbs;
  float ctr[C];
#pragma unroll
  for (int c = 0; c < C; ++c) ctr[c] = xb[c * ld + n];
  float best[KNN]; int bidx[KNN];
  for (int j = 0; j < KNN; ++j) { best[j] = -3.4e38f; bidx[j] = 0; }
  for (int m = 0; m < N; ++m) {
    float d = 0.f;
#pragma unroll
    for (int c = 0; c < C; ++c) {
      float df = xb[c * ld + m] - ctr[c];
      d = fmaf(df, df, d);
    }
    float pd = -d;
    if (pd > best[KNN - 1]) {
      int p = KNN - 1;
      while (p > 0 && best[p - 1] < pd) {
        best[p] = best[p - 1]; bidx[p] = bidx[p - 1]; --p;
      }
      best[p] = pd; bidx[p] = m;
    }
  }
  int* op = idx + ((size_t)b * N + n) * KNN;
  for (int j = 0; j < KNN; ++j) op[j] = bidx[j];
}

// ---------------------------------------------------------------------------
// Fused edge-conv block, one workgroup (4 waves) per point n:
//   edge[2C,48] built in LDS (fragment order) from idx; conv1 -> bn -> lrelu
//   [-> conv2 -> bn -> lrelu] -> max over 40 k-columns -> out[b,orow+o,n]
// W1/W2 are pre-fragmented f16 (linear 32B copies into LDS).
// ---------------------------------------------------------------------------
template <int C, int O1, int O2, bool TWO>
__global__ void __launch_bounds__(128)
edgeconv_kernel(const float* __restrict__ xin, long xbs, int xld,
                const int* __restrict__ idx,
                const f16* __restrict__ W1,
                const float* __restrict__ g1, const float* __restrict__ b1,
                const float* __restrict__ m1, const float* __restrict__ v1,
                const f16* __restrict__ W2,
                const float* __restrict__ g2, const float* __restrict__ b2,
                const float* __restrict__ m2, const float* __restrict__ v2,
                float* __restrict__ out, long obs, int oldd, int orow, int N) {
  constexpr int KP1 = ((2 * C + 31) / 32) * 32;
  constexpr int KT1 = KP1 / 32;                 // k-tiles of conv1
  __shared__ alignas(32) f16 sW1[(O1 / 16) * KT1 * FRAG];
  __shared__ alignas(32) f16 sEdge[KT1 * 3 * FRAG];
  __shared__ alignas(32) f16 sW2[TWO ? (O2 / 16) * (O1 / 32) * FRAG : 32];
  __shared__ alignas(32) f16 sH[TWO ? (O1 / 32) * 3 * FRAG : 32];
  __shared__ int   sIdx[KNN];
  __shared__ float sCtr[C];

  const int tid = threadIdx.x;
  const int n = blockIdx.x, b = blockIdx.y;
  const float* xb = xin + (size_t)b * xbs;

  // ---- phase 0: zero edge pad; linear weight copies; idx + center ----
  {
    v16h z = {};
    constexpr int NZ = KT1 * 3 * FRAG / 16;
#pragma unroll
    for (int t = tid; t < NZ; t += 128) ((v16h*)sEdge)[t] = z;
    constexpr int NV1 = (O1 / 16) * KT1 * FRAG / 16;
#pragma unroll
    for (int t = tid; t < NV1; t += 128) ((v16h*)sW1)[t] = ((const v16h*)W1)[t];
    if constexpr (TWO) {
      constexpr int NV2 = (O2 / 16) * (O1 / 32) * FRAG / 16;
#pragma unroll
      for (int t = tid; t < NV2; t += 128) ((v16h*)sW2)[t] = ((const v16h*)W2)[t];
    }
  }
  if (tid < KNN) sIdx[tid] = idx[((size_t)b * N + n) * KNN + tid];
  if (tid >= 64 && tid < 64 + C) sCtr[tid - 64] = xb[(tid - 64) * xld + n];
  __syncthreads();

  // ---- phase 1: edge features (neighbor gather batched in registers) ----
  {
    constexpr int NEL = C * KNN;
    constexpr int NIT = (NEL + 127) / 128;
    float nv[NIT];
#pragma unroll
    for (int it = 0; it < NIT; ++it) {
      int t = tid + it * 128;
      if (NEL % 128 == 0 || t < NEL) {
        int c = t / KNN, j = t % KNN;
        nv[it] = xb[c * xld + sIdx[j]];
      }
    }
#pragma unroll
    for (int it = 0; it < NIT; ++it) {
      int t = tid + it * 128;
      if (NEL % 128 == 0 || t < NEL) {
        int c = t / KNN, j = t % KNN;
        sEdge[((c >> 5) * 3 + (j >> 4)) * FRAG + fragB_pos(c & 31, j & 15)] =
            (f16)(nv[it] - sCtr[c]);
      }
    }
    // center rows [C, 2C)
#pragma unroll
    for (int it = 0; it < NIT; ++it) {
      int t = tid + it * 128;
      if (NEL % 128 == 0 || t < NEL) {
        int c = t / KNN, j = t % KNN;
        int cc = C + c;
        sEdge[((cc >> 5) * 3 + (j >> 4)) * FRAG + fragB_pos(cc & 31, j & 15)] =
            (f16)sCtr[c];
      }
    }
  }
  __syncthreads();

  const int wave = tid >> 5, lane = tid & 31;
  const int lg = (lane >> 4) & 1, ln = lane & 15;

  // conv1: O1/16 == 4 m-tiles, one per wave
  for (int mt = wave; mt < O1 / 16; mt += 4) {
    float rmax[8];
#pragma unroll
    for (int r = 0; r < 8; ++r) rmax[r] = -3.4e38f;
    for (int nt = 0; nt < 3; ++nt) {
      v8f acc = {};
#pragma unroll
      for (int kb = 0; kb < KT1; ++kb)
        acc = wmma_f16(frag_load(&sW1[(mt * KT1 + kb) * FRAG], lane),
                       frag_load(&sEdge[(kb * 3 + nt) * FRAG], lane), acc);
#pragma unroll
      for (int r = 0; r < 8; ++r) {
        int o = mt * 16 + r + 8 * lg;
        float y = lrelu(bn_apply(acc[r], g1, b1, m1, v1, o));
        if constexpr (TWO) {
          // write straight into conv2's B-fragment layout: k=o, n=ln, tile nt
          sH[((o >> 5) * 3 + nt) * FRAG + fragB_pos(o & 31, ln)] = (f16)y;
        } else {
          if (nt * 16 + ln < KNN) rmax[r] = fmaxf(rmax[r], y);
        }
      }
    }
    if constexpr (!TWO) {
#pragma unroll
      for (int r = 0; r < 8; ++r) {
        float v = rmax[r];
        v = fmaxf(v, __shfl_xor(v, 1, 32));
        v = fmaxf(v, __shfl_xor(v, 2, 32));
        v = fmaxf(v, __shfl_xor(v, 4, 32));
        v = fmaxf(v, __shfl_xor(v, 8, 32));
        if (ln == 0)
          out[(size_t)b * obs + (size_t)(orow + mt * 16 + r + 8 * lg) * oldd + n] = v;
      }
    }
  }

  if constexpr (TWO) {
    __syncthreads();
    for (int mt = wave; mt < O2 / 16; mt += 4) {
      float rmax[8];
#pragma unroll
      for (int r = 0; r < 8; ++r) rmax[r] = -3.4e38f;
      for (int nt = 0; nt < 3; ++nt) {
        v8f acc = {};
#pragma unroll
        for (int kb = 0; kb < O1 / 32; ++kb)
          acc = wmma_f16(frag_load(&sW2[(mt * (O1 / 32) + kb) * FRAG], lane),
                         frag_load(&sH[(kb * 3 + nt) * FRAG], lane), acc);
#pragma unroll
        for (int r = 0; r < 8; ++r) {
          int o = mt * 16 + r + 8 * lg;
          float y = lrelu(bn_apply(acc[r], g2, b2, m2, v2, o));
          if (nt * 16 + ln < KNN) rmax[r] = fmaxf(rmax[r], y);
        }
      }
#pragma unroll
      for (int r = 0; r < 8; ++r) {
        float v = rmax[r];
        v = fmaxf(v, __shfl_xor(v, 1, 32));
        v = fmaxf(v, __shfl_xor(v, 2, 32));
        v = fmaxf(v, __shfl_xor(v, 4, 32));
        v = fmaxf(v, __shfl_xor(v, 8, 32));
        if (ln == 0)
          out[(size_t)b * obs + (size_t)(orow + mt * 16 + r + 8 * lg) * oldd + n] = v;
      }
    }
  }
}

// ---------------------------------------------------------------------------
// WMMA GEMM + BN + LeakyReLU, out[b,o,n]. Requires Cin%32==0 and C1%32==0, so
// each 32-row K-chunk is uniformly src1 (inc1==0 => broadcast over n) or src2.
// Wh is pre-fragmented f16: [O/16][Cin/32] fragments. Double-buffered staging.
// Block: 128 threads = 4 waves = 64 M x 16 N tile.
// ---------------------------------------------------------------------------
__global__ void __launch_bounds__(128)
gemm_bn_lrelu_kernel(const float* __restrict__ src1, long s1, int ld1, int inc1, int C1,
                     const float* __restrict__ src2, long s2, int ld2,
                     int Cin,
                     const f16* __restrict__ Wh,
                     const float* __restrict__ gg, const float* __restrict__ bb,
                     const float* __restrict__ mm, const float* __restrict__ vv,
                     float* __restrict__ out, int O, int N) {
  __shared__ alignas(32) f16 sA[2][4 * FRAG];   // 64x32 weights chunk
  __shared__ alignas(32) f16 sB[2][FRAG];       // 32x16 activations chunk
  const int tid = threadIdx.x;
  const int nbase = blockIdx.x * 16, mbase = blockIdx.y * 64, b = blockIdx.z;
  const int wave = tid >> 5, lane = tid & 31;
  const int lg = (lane >> 4) & 1, ln = lane & 15;
  const int nk = Cin >> 5;

  auto stage = [&](int kb, int buf) {
    // weights: 4 fragments, one 32B vector per thread, linear copy
    {
      int mt = tid >> 5, q = tid & 31;
      ((v16h*)(sA[buf] + mt * FRAG))[q] =
          ((const v16h*)(Wh + ((size_t)((mbase >> 4) + mt) * nk + kb) * FRAG))[q];
    }
    // activations: 4 elements per thread, load-phase then store-phase
    const int gr = kb * 32;
    float bv[4];
    if (gr < C1) {
#pragma unroll
      for (int it = 0; it < 4; ++it) {
        int t = tid + it * 128;
        int c = t >> 4, nn = t & 15;
        bv[it] = src1[(size_t)b * s1 + (size_t)(gr + c) * ld1 +
                      (size_t)(nbase + nn) * inc1];
      }
    } else {
#pragma unroll
      for (int it = 0; it < 4; ++it) {
        int t = tid + it * 128;
        int c = t >> 4, nn = t & 15;
        bv[it] = src2[(size_t)b * s2 + (size_t)(gr - C1 + c) * ld2 + nbase + nn];
      }
    }
#pragma unroll
    for (int it = 0; it < 4; ++it) {
      int t = tid + it * 128;
      int c = t >> 4, nn = t & 15;
      sB[buf][fragB_pos(c, nn)] = (f16)bv[it];
    }
  };

  v8f acc = {};
  stage(0, 0);
  __syncthreads();
  for (int kb = 0; kb < nk; ++kb) {
    int cur = kb & 1;
    if (kb + 1 < nk) stage(kb + 1, cur ^ 1);
    acc = wmma_f16(frag_load(&sA[cur][wave * FRAG], lane),
                   frag_load(sB[cur], lane), acc);
    __syncthreads();
  }
#pragma unroll
  for (int r = 0; r < 8; ++r) {
    int o = mbase + wave * 16 + r + 8 * lg;
    float y = lrelu(bn_apply(acc[r], gg, bb, mm, vv, o));
    out[((size_t)b * O + o) * N + nbase + ln] = y;
  }
}

// ---------------------------------------------------------------------------
// WMMA GEMM + BN + LeakyReLU + max over all N columns -> outmax[b,o].
// Each block exclusively owns 64 output channels => deterministic, no atomics.
// CIN compile-time (128/192): weights resident in LDS (linear copy of
// pre-fragmented f16); activations double-buffered, register-batched staging.
// ---------------------------------------------------------------------------
template <int CIN>
__global__ void __launch_bounds__(128)
gemm_max_kernel(const float* __restrict__ src, long sbs,
                const f16* __restrict__ Wh,
                const float* __restrict__ gg, const float* __restrict__ bb,
                const float* __restrict__ mm, const float* __restrict__ vv,
                float* __restrict__ outmax, int O, int N) {
  constexpr int NK = CIN / 32;
  __shared__ alignas(32) f16 sW[4 * NK * FRAG]; // 64 x CIN weights
  __shared__ alignas(32) f16 sX[2][NK * FRAG];  // CIN x 16 activations
  const int tid = threadIdx.x;
  const int mbase = blockIdx.x * 64, b = blockIdx.y;
  const int wave = tid >> 5, lane = tid & 31;
  const int lg = (lane >> 4) & 1, ln = lane & 15;

  {
    constexpr int NV = 4 * NK * 32;  // v16h count
    const v16h* srcW = (const v16h*)Wh + (size_t)(mbase >> 4) * NK * 32;
#pragma unroll
    for (int t = tid; t < NV; t += 128) ((v16h*)sW)[t] = srcW[t];
  }

  auto stageX = [&](int nc, int buf) {
    constexpr int NXIT = CIN * 16 / 128;
    float xv[NXIT];
#pragma unroll
    for (int it = 0; it < NXIT; ++it) {
      int t = tid + it * 128;
      int c = t >> 4, nn = t & 15;
      xv[it] = src[(size_t)b * sbs + (size_t)c * N + nc * 16 + nn];
    }
#pragma unroll
    for (int it = 0; it < NXIT; ++it) {
      int t = tid + it * 128;
      int c = t >> 4, nn = t & 15;
      sX[buf][(c >> 5) * FRAG + fragB_pos(c & 31, nn)] = (f16)xv[it];
    }
  };

  float rmax[8];
#pragma unroll
  for (int r = 0; r < 8; ++r) rmax[r] = -3.4e38f;

  stageX(0, 0);
  __syncthreads();
  const int NC = N / 16;
  for (int nc = 0; nc < NC; ++nc) {
    int cur = nc & 1;
    if (nc + 1 < NC) stageX(nc + 1, cur ^ 1);
    v8f acc = {};
#pragma unroll
    for (int kb = 0; kb < NK; ++kb)
      acc = wmma_f16(frag_load(&sW[(wave * NK + kb) * FRAG], lane),
                     frag_load(&sX[cur][kb * FRAG], lane), acc);
#pragma unroll
    for (int r = 0; r < 8; ++r) {
      int o = mbase + wave * 16 + r + 8 * lg;
      rmax[r] = fmaxf(rmax[r], lrelu(bn_apply(acc[r], gg, bb, mm, vv, o)));
    }
    __syncthreads();
  }
#pragma unroll
  for (int r = 0; r < 8; ++r) {
    float v = rmax[r];
    v = fmaxf(v, __shfl_xor(v, 1, 32));
    v = fmaxf(v, __shfl_xor(v, 2, 32));
    v = fmaxf(v, __shfl_xor(v, 4, 32));
    v = fmaxf(v, __shfl_xor(v, 8, 32));
    if (ln == 0) outmax[(size_t)b * O + mbase + wave * 16 + r + 8 * lg] = v;
  }
}

// ---------------------------------------------------------------------------
// Small FC for the transform-net tail (B=8; tiny).
// ---------------------------------------------------------------------------
__global__ void fc_kernel(const float* __restrict__ in, int Cin,
                          const float* __restrict__ W, int O,
                          const float* __restrict__ gg, const float* __restrict__ bb,
                          const float* __restrict__ mm, const float* __restrict__ vv,
                          const float* __restrict__ bias, int do_lrelu,
                          float* __restrict__ out) {
  int o = blockIdx.x * blockDim.x + threadIdx.x;
  int b = blockIdx.y;
  if (o >= O) return;
  const float* ib = in + (size_t)b * Cin;
  const float* wr = W + (size_t)o * Cin;
  float acc = 0.f;
  for (int c = 0; c < Cin; ++c) acc = fmaf(ib[c], wr[c], acc);
  if (bias) acc += bias[o];
  if (gg) acc = bn_apply(acc, gg, bb, mm, vv, o);
  if (do_lrelu) acc = lrelu(acc);
  out[(size_t)b * O + o] = acc;
}

// xT[b,d,n] = sum_c x[b,c,n] * t[b, 3c+d]
__global__ void transform_kernel(const float* __restrict__ x, const float* __restrict__ t,
                                 float* __restrict__ xT, int N) {
  int n = blockIdx.x * blockDim.x + threadIdx.x;
  int b = blockIdx.y;
  if (n >= N) return;
  const float* xb = x + (size_t)b * 3 * N;
  const float* tb = t + b * 9;
  float x0 = xb[0 * N + n], x1 = xb[1 * N + n], x2 = xb[2 * N + n];
  float* ob = xT + (size_t)b * 3 * N;
#pragma unroll
  for (int d = 0; d < 3; ++d)
    ob[d * N + n] = x0 * tb[d] + x1 * tb[3 + d] + x2 * tb[6 + d];
}

// conv11 (3x128) + sigmoid*2-1 + L2 normalize; out[b,n,3]
__global__ void head_kernel(const float* __restrict__ h, const float* __restrict__ w,
                            float* __restrict__ out, int N) {
  int n = blockIdx.x * blockDim.x + threadIdx.x;
  int b = blockIdx.y;
  if (n >= N) return;
  const float* hb = h + (size_t)b * 128 * N;
  float y[3];
#pragma unroll
  for (int d = 0; d < 3; ++d) {
    float acc = 0.f;
    for (int c = 0; c < 128; ++c) acc = fmaf(hb[c * N + n], w[d * 128 + c], acc);
    y[d] = 2.f / (1.f + __expf(-acc)) - 1.f;
  }
  float nrm = sqrtf(y[0] * y[0] + y[1] * y[1] + y[2] * y[2]);
  float inv = 1.f / fmaxf(nrm, 1e-12f);
  float* ob = out + ((size_t)b * N + n) * 3;
  ob[0] = y[0] * inv; ob[1] = y[1] * inv; ob[2] = y[2] * inv;
}

// ---------------------------------------------------------------------------
extern "C" void kernel_launch(void* const* d_in, const int* in_sizes, int n_in,
                              void* d_out, int out_size, void* d_ws, size_t ws_size,
                              hipStream_t stream) {
  (void)in_sizes; (void)n_in; (void)out_size; (void)ws_size;
  const int B = 8, N = 2048;
  auto F = [&](int i) { return (const float*)d_in[i]; };

  // setup_inputs() flattening order (nested bn dicts in g,b,m,v order)
  const float* x_in       = F(0);
  const float* t_conv1_w  = F(2);
  const float *tb1g=F(3),*tb1b=F(4),*tb1m=F(5),*tb1v=F(6);
  const float* t_conv2_w  = F(7);
  const float *tb2g=F(8),*tb2b=F(9),*tb2m=F(10),*tb2v=F(11);
  const float* t_conv3_w  = F(12);
  const float *tb3ag=F(13),*tb3ab=F(14),*tb3am=F(15),*tb3av=F(16);
  const float* t_lin1_w   = F(17);
  const float *tb3g=F(18),*tb3b=F(19),*tb3m=F(20),*tb3v=F(21);
  const float* t_lin2_w   = F(22);
  const float *tb4g=F(23),*tb4b=F(24),*tb4m=F(25),*tb4v=F(26);
  const float* t_trans_w  = F(27);
  const float* t_trans_b  = F(28);
  const float* conv1_w    = F(29);
  const float *b1g=F(30),*b1b=F(31),*b1m=F(32),*b1v=F(33);
  const float* conv2_w    = F(34);
  const float *b2g=F(35),*b2b=F(36),*b2m=F(37),*b2v=F(38);
  const float* conv3_w    = F(39);
  const float *b3g=F(40),*b3b=F(41),*b3m=F(42),*b3v=F(43);
  const float* conv4_w    = F(44);
  const float *b4g=F(45),*b4b=F(46),*b4m=F(47),*b4v=F(48);
  const float* conv5_w    = F(49);
  const float *b5g=F(50),*b5b=F(51),*b5m=F(52),*b5v=F(53);
  const float* conv6_w    = F(54);
  const float *b6g=F(55),*b6b=F(56),*b6m=F(57),*b6v=F(58);
  const float* conv8_w    = F(59);
  const float *b8g=F(60),*b8b=F(61),*b8m=F(62),*b8v=F(63);
  const float* conv9_w    = F(64);
  const float *b9g=F(65),*b9b=F(66),*b9m=F(67),*b9v=F(68);
  const float* conv10_w   = F(69);
  const float *b10g=F(70),*b10b=F(71),*b10m=F(72),*b10v=F(73);
  const float* conv11_w   = F(74);

  // workspace layout (~65 MB), every block 32B-aligned
  float* ws = (float*)d_ws;
  size_t off = 0;
  auto alloc = [&](size_t nfloats) {
    float* p = ws + off;
    off = (off + nfloats + 7) & ~(size_t)7;
    return p;
  };
  int*   idxb  = (int*)alloc((size_t)B * N * KNN);   // reused per kNN stage
  float* xT    = alloc((size_t)B * 3 * N);
  float* th128 = alloc((size_t)B * 128 * N);
  float* x123  = alloc((size_t)B * 192 * N);         // [x1;x2;x3]
  float* tg    = alloc((size_t)B * 1024);
  float* tf1   = alloc((size_t)B * 512);
  float* tf2   = alloc((size_t)B * 256);
  float* tmat  = alloc((size_t)B * 9);
  float* gfeat = alloc((size_t)B * 1024);
  float* ha    = alloc((size_t)B * 256 * N);
  float* hb    = alloc((size_t)B * 256 * N);
  float* hc    = alloc((size_t)B * 128 * N);

  // pre-fragmented f16 weights (1 KB fragments, 32B aligned)
  f16* whbase = (f16*)alloc(774144 / 2 + 8);
  size_t wo = 0;
  auto walloc = [&](size_t n) { f16* p = whbase + wo; wo += n; return p; };
  f16* w_tc1 = walloc(64 * 32);
  f16* w_tc2 = walloc(128 * 64);
  f16* w_tc3 = walloc(1024 * 128);
  f16* w_c1  = walloc(64 * 32);
  f16* w_c2  = walloc(64 * 64);
  f16* w_c3  = walloc(64 * 128);
  f16* w_c4  = walloc(64 * 64);
  f16* w_c5  = walloc(64 * 128);
  f16* w_c6  = walloc(1024 * 192);
  f16* w_c8  = walloc(256 * 1216);
  f16* w_c9  = walloc(256 * 256);
  f16* w_c10 = walloc(128 * 256);

  auto prep = [&](const float* W, int O, int Cin, int KP, f16* dst) {
    int total = O * KP;
    prep_weights_kernel<<<(total + 255) / 256, 256, 0, stream>>>(W, O, Cin, KP, dst);
  };
  prep(t_conv1_w, 64, 6, 32, w_tc1);
  prep(t_conv2_w, 128, 64, 64, w_tc2);
  prep(t_conv3_w, 1024, 128, 128, w_tc3);
  prep(conv1_w, 64, 6, 32, w_c1);
  prep(conv2_w, 64, 64, 64, w_c2);
  prep(conv3_w, 64, 128, 128, w_c3);
  prep(conv4_w, 64, 64, 64, w_c4);
  prep(conv5_w, 64, 128, 128, w_c5);
  prep(conv6_w, 1024, 192, 192, w_c6);
  prep(conv8_w, 256, 1216, 1216, w_c8);
  prep(conv9_w, 256, 256, 256, w_c9);
  prep(conv10_w, 128, 256, 256, w_c10);

  dim3 gPT(N / 256, B);   // per-point scalar kernels
  dim3 gEC(N, B);         // edge-conv: one block per point

  // ---- transform net ----
  knn_kernel<3><<<gPT, 256, 0, stream>>>(x_in, 3L * N, N, idxb, N);
  edgeconv_kernel<3, 64, 128, true><<<gEC, 128, 0, stream>>>(
      x_in, 3L * N, N, idxb,
      w_tc1, tb1g, tb1b, tb1m, tb1v,
      w_tc2, tb2g, tb2b, tb2m, tb2v,
      th128, 128L * N, N, 0, N);
  gemm_max_kernel<128><<<dim3(1024 / 64, B), 128, 0, stream>>>(
      th128, 128L * N, w_tc3, tb3ag, tb3ab, tb3am, tb3av, tg, 1024, N);
  fc_kernel<<<dim3(4, B), 128, 0, stream>>>(tg, 1024, t_lin1_w, 512,
      tb3g, tb3b, tb3m, tb3v, nullptr, 1, tf1);
  fc_kernel<<<dim3(2, B), 128, 0, stream>>>(tf1, 512, t_lin2_w, 256,
      tb4g, tb4b, tb4m, tb4v, nullptr, 1, tf2);
  fc_kernel<<<dim3(1, B), 128, 0, stream>>>(tf2, 256, t_trans_w, 9,
      nullptr, nullptr, nullptr, nullptr, t_trans_b, 0, tmat);
  transform_kernel<<<gPT, 256, 0, stream>>>(x_in, tmat, xT, N);

  // ---- trunk ----
  knn_kernel<3><<<gPT, 256, 0, stream>>>(xT, 3L * N, N, idxb, N);
  edgeconv_kernel<3, 64, 64, true><<<gEC, 128, 0, stream>>>(
      xT, 3L * N, N, idxb,
      w_c1, b1g, b1b, b1m, b1v,
      w_c2, b2g, b2b, b2m, b2v,
      x123, 192L * N, N, 0, N);                                   // -> x1 rows 0..63

  knn_kernel<64><<<gPT, 256, 0, stream>>>(x123, 192L * N, N, idxb, N);
  edgeconv_kernel<64, 64, 64, true><<<gEC, 128, 0, stream>>>(
      x123, 192L * N, N, idxb,
      w_c3, b3g, b3b, b3m, b3v,
      w_c4, b4g, b4b, b4m, b4v,
      x123, 192L * N, N, 64, N);                                  // -> x2 rows 64..127

  knn_kernel<64><<<gPT, 256, 0, stream>>>(x123 + 64 * N, 192L * N, N, idxb, N);
  edgeconv_kernel<64, 64, 64, false><<<gEC, 128, 0, stream>>>(
      x123 + 64 * N, 192L * N, N, idxb,
      w_c5, b5g, b5b, b5m, b5v,
      nullptr, nullptr, nullptr, nullptr, nullptr,
      x123, 192L * N, N, 128, N);                                 // -> x3 rows 128..191

  gemm_max_kernel<192><<<dim3(1024 / 64, B), 128, 0, stream>>>(
      x123, 192L * N, w_c6, b6g, b6b, b6m, b6v, gfeat, 1024, N);

  // conv8: virtual concat [g(1024, broadcast); x123(192)]
  gemm_bn_lrelu_kernel<<<dim3(N / 16, 256 / 64, B), 128, 0, stream>>>(
      gfeat, 1024L, 1, 0, 1024, x123, 192L * N, N, 1216,
      w_c8, b8g, b8b, b8m, b8v, ha, 256, N);
  gemm_bn_lrelu_kernel<<<dim3(N / 16, 256 / 64, B), 128, 0, stream>>>(
      ha, 256L * N, N, 1, 256, nullptr, 0, 0, 256,
      w_c9, b9g, b9b, b9m, b9v, hb, 256, N);
  gemm_bn_lrelu_kernel<<<dim3(N / 16, 128 / 64, B), 128, 0, stream>>>(
      hb, 256L * N, N, 1, 256, nullptr, 0, 0, 256,
      w_c10, b10g, b10b, b10m, b10v, hc, 128, N);

  head_kernel<<<gPT, 256, 0, stream>>>(hc, conv11_w, (float*)d_out, N);
}
